// Codebook_13709535608878
// MI455X (gfx1250) — matrix operand order repaired and checked by reading, not compiled, not run on hardware
//
#include <hip/hip_runtime.h>
#include <hip/hip_bf16.h>
#include <math.h>

// ---------------------------------------------------------------------------
// Problem constants (from reference):
//   z:      [16, 64, 32, 32] f32   -> N_TOK = 16384 tokens of dim D = 64
//   weight: [8192, 64] f32         -> K_CODES = 8192
// Output layout (flat concat, f32):
//   z_q_out [16,64,32,32] : off 0,        1048576 elems
//   encoding_indices      : off 1048576,  16384  elems (written as float)
//   loss                  : off 1064960,  1 elem
//   encodings one-hot     : off 1064961,  134217728 elems
//   perplexity            : off 135282689, 1 elem
// ---------------------------------------------------------------------------

#define N_TOK   16384
#define D_DIM   64
#define K_CODES 8192

#define ZQ_OFF   0
#define IDX_OFF  1048576
#define LOSS_OFF 1064960
#define ENC_OFF  1064961ull
#define PPL_OFF  135282689ull

typedef __attribute__((ext_vector_type(16))) __bf16         v16bf;
typedef __attribute__((ext_vector_type(8)))  float          v8f;
typedef __attribute__((ext_vector_type(8)))  unsigned short ushort8;

__device__ __forceinline__ unsigned short f32_to_bf16(float f) {
    unsigned int u = __float_as_uint(f);
    unsigned int r = (u + 0x7FFFu + ((u >> 16) & 1u)) >> 16;   // round-nearest-even
    return (unsigned short)r;
}

// ---------------------------------------------------------------------------
// K1: L2-normalize codebook rows -> f32 copy (for gather) + bf16 copy (for WMMA)
// ---------------------------------------------------------------------------
__global__ __launch_bounds__(256) void norm_weight_kernel(
    const float* __restrict__ w, float* __restrict__ wn,
    unsigned short* __restrict__ wn_bf16)
{
    const int lane = threadIdx.x & 31;
    const int wave = threadIdx.x >> 5;
    const int row  = blockIdx.x * 8 + wave;          // 8192 rows, grid 1024
    const float x0 = w[row * 64 + lane];
    const float x1 = w[row * 64 + lane + 32];
    float ss = x0 * x0 + x1 * x1;
#pragma unroll
    for (int m = 16; m >= 1; m >>= 1) ss += __shfl_xor(ss, m, 32);
    const float rn = 1.0f / fmaxf(sqrtf(ss), 1e-12f);
    const float y0 = x0 * rn, y1 = x1 * rn;
    wn[row * 64 + lane]           = y0;
    wn[row * 64 + lane + 32]      = y1;
    wn_bf16[row * 64 + lane]      = f32_to_bf16(y0);
    wn_bf16[row * 64 + lane + 32] = f32_to_bf16(y1);
}

// ---------------------------------------------------------------------------
// K2: BHWC-transpose + L2-normalize tokens via padded-LDS transpose.
// ---------------------------------------------------------------------------
__global__ __launch_bounds__(256) void norm_z_kernel(
    const float* __restrict__ z, float* __restrict__ zf,
    unsigned short* __restrict__ zf_bf16)
{
    __shared__ float lds[32 * 65];                   // [w][c], pad 65 vs 64 banks
    const int t  = threadIdx.x;
    const int bh = blockIdx.x;                       // b*32 + h  (grid 512)
    const int b  = bh >> 5, h = bh & 31;
    const size_t base = (size_t)b * 65536 + (size_t)h * 32;
    const int wq = t & 31;                           // w
    const int c0 = t >> 5;                           // 0..7
#pragma unroll
    for (int j = 0; j < 8; ++j) {
        const int c = c0 + j * 8;
        lds[wq * 65 + c] = z[base + (size_t)c * 1024 + wq];
    }
    __syncthreads();
    const int tok = t >> 3;                          // w index 0..31
    const int p   = t & 7;                           // 8 threads per token
    float ss = 0.f;
#pragma unroll
    for (int j = 0; j < 8; ++j) { const float v = lds[tok * 65 + p * 8 + j]; ss += v * v; }
    ss += __shfl_xor(ss, 1, 32);
    ss += __shfl_xor(ss, 2, 32);
    ss += __shfl_xor(ss, 4, 32);
    const float rn = 1.0f / fmaxf(sqrtf(ss), 1e-12f);
    const size_t n = (size_t)bh * 32 + tok;          // token id: ((b*32+h)*32+w)
#pragma unroll
    for (int j = 0; j < 8; ++j) {
        const float v = lds[tok * 65 + p * 8 + j] * rn;
        zf[n * 64 + p * 8 + j]      = v;
        zf_bf16[n * 64 + p * 8 + j] = f32_to_bf16(v);
    }
}

// ---------------------------------------------------------------------------
// K3: fused cosine-GEMM + argmax via V_WMMA_F32_16X16X32_BF16, with the code
// tiles double-buffered in LDS through the CDNA5 async-copy path
// (GLOBAL_LOAD_ASYNC_TO_LDS_B64 tracked by ASYNCcnt, §08_async_tensor).
//
// Mapping: A = 16 codes x 64 dims, B = 64 dims x 16 tokens, C[M=code][N=token].
// C layout (32b 16x16): lane = token column, 8 VGPRs = 8 code rows, so the
// argmax over codes is a per-lane register op; one __shfl_xor(16) merges the
// two row halves at the end.
//
// All 8 waves of a block sweep the SAME 512 code tiles in lockstep; the block
// cooperatively async-copies tile ct+1 (2KB) into LDS while tile ct feeds the
// WMMAs (8x less L2 traffic than per-wave fetching). LDS rows padded to 144B
// (36 words; 16 distinct banks across the 16 row-readers, 16B-aligned rows so
// fragment reads are ds_read_b128).
// ---------------------------------------------------------------------------
__global__ __launch_bounds__(256) void argmax_wmma_kernel(
    const unsigned short* __restrict__ wn_bf16,
    const unsigned short* __restrict__ zf_bf16,
    int* __restrict__ idx_out)
{
    __shared__ ushort8 abuf[2][16][9];               // 2 x 16 rows x 144B

    const int t    = threadIdx.x;
    const int lane = t & 31;
    const int wave = t >> 5;
    const int tile = blockIdx.x * 8 + wave;          // 0..1023 token tiles
    const int t0   = tile * 16;
    const int tcol = lane & 15;
    const bool hi  = lane >= 16;

    // ---- B (tokens): register-resident for the whole sweep. -------------
    // Lane holds column N=tcol; lanes 0-15 K=0..7,16..23, lanes 16-31 K=8..15,24..31.
    const int koff_b = hi ? 8 : 0;
    const unsigned short* bp = zf_bf16 + (size_t)(t0 + tcol) * 64;
    union { v16bf v; ushort8 u[2]; } B0, B1;
    B0.u[0] = *(const ushort8*)(bp + koff_b);
    B0.u[1] = *(const ushort8*)(bp + koff_b + 16);
    B1.u[0] = *(const ushort8*)(bp + 32 + koff_b);
    B1.u[1] = *(const ushort8*)(bp + 32 + koff_b + 16);

    // ---- cooperative async fill: thread t moves 8B of a 16x128B tile ----
    const int frow = t >> 4;                         // 0..15 (code row in tile)
    const int fseg = t & 15;                         // 0..15 (8B segment)

    #define ISSUE_FILL(ct_, buf_)                                              \
    do {                                                                       \
        unsigned long long g_ = (unsigned long long)(uintptr_t)(               \
            wn_bf16 + ((size_t)(ct_) * 16 + frow) * 64 + fseg * 4);            \
        unsigned int l_ = (unsigned int)(uintptr_t)(                           \
            (char*)&abuf[(buf_)][frow][0] + fseg * 8);                         \
        asm volatile("global_load_async_to_lds_b64 %0, %1, off"                \
                     :: "v"(l_), "v"(g_) : "memory");                          \
    } while (0)

    ISSUE_FILL(0, 0);                                // prologue: tile 0 -> buf 0

    const int sel = hi ? 1 : 0;                      // A K-half select per lane
    float best = -2.0f;                              // cosine in [-1,1]
    int   bidx = 0;

    for (int ct = 0; ct < 512; ++ct) {
        const int cur = ct & 1;
        // Keep exactly one fill in flight: issue ct+1 (wrapped; last one is a
        // harmless refetch of tile 0) so "s_wait_asynccnt 1" always retires
        // the fill for tile ct.
        ISSUE_FILL((ct + 1) & 511, cur ^ 1);
        asm volatile("s_wait_asynccnt 0x1" ::: "memory");
        __syncthreads();                             // tile ct visible to all waves

        // A fragments from LDS: lane reads row M=tcol, 4x ds_read_b128.
        const ushort8* row = &abuf[cur][tcol][0];
        union { v16bf v; ushort8 u[2]; } A0, A1;
        A0.u[0] = row[0 + sel];                      // K {0-7 | 8-15}
        A0.u[1] = row[2 + sel];                      // K {16-23 | 24-31}
        A1.u[0] = row[4 + sel];                      // K {32-39 | 40-47}
        A1.u[1] = row[6 + sel];                      // K {48-55 | 56-63}

        v8f c = {};
        c = __builtin_amdgcn_wmma_f32_16x16x32_bf16(false, A0.v, false, B0.v,
                                                    (short)0, c, false, false);
        c = __builtin_amdgcn_wmma_f32_16x16x32_bf16(false, A1.v, false, B1.v,
                                                    (short)0, c, false, false);

        // Tree argmax over the 8 accumulator rows (depth-3 chain instead of 8).
        // Strict '>' keeps the lowest index on ties (first-max semantics).
        float m0 = c[0]; int i0 = 0;
        if (c[1] > m0) { m0 = c[1]; i0 = 1; }
        float m1 = c[2]; int i1 = 2;
        if (c[3] > m1) { m1 = c[3]; i1 = 3; }
        float m2 = c[4]; int i2 = 4;
        if (c[5] > m2) { m2 = c[5]; i2 = 5; }
        float m3 = c[6]; int i3 = 6;
        if (c[7] > m3) { m3 = c[7]; i3 = 7; }
        if (m1 > m0) { m0 = m1; i0 = i1; }
        if (m3 > m2) { m2 = m3; i2 = i3; }
        if (m2 > m0) { m0 = m2; i0 = i2; }
        const int mbase = ct * 16 + (hi ? 8 : 0);
        if (m0 > best) { best = m0; bidx = mbase + i0; }

        __syncthreads();                             // buf[cur] free for refill
    }
    #undef ISSUE_FILL

    // Merge lane l (codes ==0..7 mod 16) with lane l+16 (codes ==8..15 mod 16).
    const float ov = __shfl_xor(best, 16, 32);
    const int   oi = __shfl_xor(bidx, 16, 32);
    if (ov > best || (ov == best && oi < bidx)) { best = ov; bidx = oi; }
    if (!hi) idx_out[t0 + tcol] = bidx;
}

// ---------------------------------------------------------------------------
// K4: scatter indices (as float), one-hot encodings, and code histogram.
// ---------------------------------------------------------------------------
__global__ __launch_bounds__(256) void scatter_kernel(
    const int* __restrict__ idx, float* __restrict__ dout,
    unsigned int* __restrict__ counts)
{
    const int n  = blockIdx.x * 256 + threadIdx.x;   // < 16384
    const int id = idx[n];
    dout[IDX_OFF + n] = (float)id;
    dout[ENC_OFF + (size_t)n * K_CODES + (size_t)id] = 1.0f;
    atomicAdd(&counts[id], 1u);
}

// ---------------------------------------------------------------------------
// K5: z_q gather back to BCHW + fused (1+beta)*MSE loss reduction.
// ---------------------------------------------------------------------------
__global__ __launch_bounds__(256) void zq_loss_kernel(
    const float* __restrict__ wn, const float* __restrict__ zf,
    const int* __restrict__ idx, float* __restrict__ dout)
{
    __shared__ float red[8];
    const int o = blockIdx.x * 256 + threadIdx.x;    // < 1048576
    const int c = (o >> 10) & 63;
    const int n = ((o >> 16) << 10) | (o & 1023);    // b*1024 + h*32 + w
    const float q = wn[(size_t)idx[n] * 64 + c];
    const float r = zf[(size_t)n * 64 + c];
    dout[ZQ_OFF + o] = q;                            // straight-through == zq
    const float d = q - r;
    float acc = d * d;
#pragma unroll
    for (int m = 16; m >= 1; m >>= 1) acc += __shfl_xor(acc, m, 32);
    if ((threadIdx.x & 31) == 0) red[threadIdx.x >> 5] = acc;
    __syncthreads();
    if (threadIdx.x == 0) {
        float s = 0.f;
#pragma unroll
        for (int i = 0; i < 8; ++i) s += red[i];
        atomicAdd(dout + LOSS_OFF, s * (1.01f / 1048576.0f));  // (1+beta)*mean
    }
}

// ---------------------------------------------------------------------------
// K6: perplexity from histogram (single block).
// ---------------------------------------------------------------------------
__global__ __launch_bounds__(256) void perplexity_kernel(
    const unsigned int* __restrict__ counts, float* __restrict__ dout)
{
    __shared__ float red[8];
    float s = 0.f;
    for (int k = threadIdx.x; k < K_CODES; k += 256) {
        const float p = (float)counts[k] * (1.0f / 16384.0f);
        s += p * logf(p + 1e-10f);
    }
#pragma unroll
    for (int m = 16; m >= 1; m >>= 1) s += __shfl_xor(s, m, 32);
    if ((threadIdx.x & 31) == 0) red[threadIdx.x >> 5] = s;
    __syncthreads();
    if (threadIdx.x == 0) {
        float t = 0.f;
#pragma unroll
        for (int i = 0; i < 8; ++i) t += red[i];
        dout[PPL_OFF] = expf(-t);
    }
}

// ---------------------------------------------------------------------------
// Workspace layout (bytes): wn_f32 2MB | zf_f32 4MB | wn_bf16 1MB | zf_bf16 2MB
//                           | idx 64KB | counts 32KB   (~9.1 MB total)
// ---------------------------------------------------------------------------
extern "C" void kernel_launch(void* const* d_in, const int* in_sizes, int n_in,
                              void* d_out, int out_size, void* d_ws, size_t ws_size,
                              hipStream_t stream)
{
    const float* z = (const float*)d_in[0];          // [16,64,32,32]
    const float* w = (const float*)d_in[1];          // [8192,64]
    float* dout = (float*)d_out;
    char*  ws   = (char*)d_ws;

    float*          wn_f32  = (float*)(ws + 0);
    float*          zf_f32  = (float*)(ws + 2097152);
    unsigned short* wn_bf16 = (unsigned short*)(ws + 6291456);
    unsigned short* zf_bf16 = (unsigned short*)(ws + 7340032);
    int*            idx     = (int*)(ws + 9437184);
    unsigned int*   counts  = (unsigned int*)(ws + 9502720);

    // Zero loss slot + one-hot encodings region (graph-capturable memset nodes).
    hipMemsetAsync(dout + LOSS_OFF, 0, (1ull + 134217728ull) * sizeof(float), stream);
    hipMemsetAsync(counts, 0, K_CODES * sizeof(unsigned int), stream);

    hipLaunchKernelGGL(norm_weight_kernel, dim3(1024), dim3(256), 0, stream,
                       w, wn_f32, wn_bf16);
    hipLaunchKernelGGL(norm_z_kernel, dim3(512), dim3(256), 0, stream,
                       z, zf_f32, zf_bf16);
    hipLaunchKernelGGL(argmax_wmma_kernel, dim3(128), dim3(256), 0, stream,
                       wn_bf16, zf_bf16, idx);
    hipLaunchKernelGGL(scatter_kernel, dim3(64), dim3(256), 0, stream,
                       idx, dout, counts);
    hipLaunchKernelGGL(zq_loss_kernel, dim3(4096), dim3(256), 0, stream,
                       wn_f32, zf_f32, idx, dout);
    hipLaunchKernelGGL(perplexity_kernel, dim3(1), dim3(256), 0, stream,
                       counts, dout);
}
